// S_Mamba_40381282517116
// MI455X (gfx1250) — matrix operand order repaired and checked by reading, not compiled, not run on hardware
//
#include <hip/hip_runtime.h>
#include <hip/hip_bf16.h>
#include <math.h>
#include <stdint.h>

typedef __attribute__((ext_vector_type(16))) _Float16 v16h;
typedef __attribute__((ext_vector_type(8)))  float    v8f;

#define NBATCH 32
#define LHIST  96
#define NVARS  862
#define DMODEL 512
#define DINNER 512
#define DSTATE 32
#define DTRANK 32
#define BNROWS (NBATCH*NVARS)   /* 27584 = 431*64 */
#define EPSF   1e-5f

__device__ __forceinline__ float sigmoidf_(float x){ return 1.0f/(1.0f+__expf(-x)); }
__device__ __forceinline__ float siluf_(float x){ return x*sigmoidf_(x); }

// ---------------------------------------------------------------------------
// Per-(b,n) mean/std over L=96; writes normalized transposed series as f16
// [BN,96] (embedding GEMM operand). One wave32 per series.
// ---------------------------------------------------------------------------
__global__ __launch_bounds__(256)
void stats_kernel(const float* __restrict__ hist, _Float16* __restrict__ xnh,
                  float* __restrict__ means, float* __restrict__ stdev)
{
  int wid  = blockIdx.x*8 + (threadIdx.x>>5);
  int lane = threadIdx.x & 31;
  if (wid >= BNROWS) return;
  int b = wid / NVARS, n = wid % NVARS;
  const float* base = hist + (size_t)b*LHIST*NVARS + n;
  float v0 = base[(size_t)(lane     )*NVARS];
  float v1 = base[(size_t)(lane + 32)*NVARS];
  float v2 = base[(size_t)(lane + 64)*NVARS];
  float s  = v0+v1+v2;
  float sq = v0*v0 + v1*v1 + v2*v2;
  for (int off=16; off; off>>=1){ s += __shfl_xor(s, off, 32); sq += __shfl_xor(sq, off, 32); }
  float mean = s * (1.0f/96.0f);
  float var  = sq * (1.0f/96.0f) - mean*mean;
  float sd   = sqrtf(var + EPSF);
  float inv  = 1.0f/sd;
  if (lane == 0){ means[wid] = mean; stdev[wid] = sd; }
  _Float16* row = xnh + (size_t)wid*96;
  row[lane     ] = (_Float16)((v0-mean)*inv);
  row[lane + 32] = (_Float16)((v1-mean)*inv);
  row[lane + 64] = (_Float16)((v2-mean)*inv);
}

// Weights f32 [K,N] -> transposed, zero-padded f16 [Npad,K]
__global__ __launch_bounds__(256)
void pack_w_kernel(const float* __restrict__ W, int N, int K,
                   _Float16* __restrict__ Wt, int Npad)
{
  size_t idx = (size_t)blockIdx.x*256 + threadIdx.x;
  if (idx >= (size_t)Npad*K) return;
  size_t n = idx / (size_t)K;
  int    k = (int)(idx - n*(size_t)K);
  Wt[idx] = (n < (size_t)N) ? (_Float16)W[(size_t)k*N + n] : (_Float16)0.0f;
}

// ---------------------------------------------------------------------------
// f16-WMMA GEMM: act(Ah[M,K] @ Wt[N,K]^T + bias) -> optional f32 C (=/+=) and
// optional f16 Ch (feeds the next GEMM). Block: 128 thr (4 waves), 64x128 tile,
// K-step 32. Double-buffered LDS tiles staged with global_load_async_to_lds_b128
// (ASYNCcnt pipeline: 6 in-flight copies per thread per tile, in-order done).
// act: 0 none, 1 relu, 2 softplus
// ---------------------------------------------------------------------------
#define GBM 64
#define GBN 128
#define GBK 32
#define KP  (GBK + 8)   /* padded row, halves */

__global__ __launch_bounds__(128)
void gemm_kernel(const _Float16* __restrict__ Ah, int ldaH,
                 const _Float16* __restrict__ Wt,
                 const float* __restrict__ bias,
                 float* __restrict__ C, int ldc,
                 _Float16* __restrict__ Ch, int ldh,
                 int N, int K, int act, int accum)
{
  __shared__ _Float16 As[2][GBM][KP];   // [buf][row][k]
  __shared__ _Float16 Ws[2][GBN][KP];   // [buf][col][k]

  const int tid  = threadIdx.x;
  const int lane = tid & 31;
  const int wv   = tid >> 5;
  const int wm   = wv >> 1;          // 0..1 -> 32-row strip
  const int wn   = wv & 1;           // 0..1 -> 64-col strip
  const int hi   = lane >> 4;
  const int l16  = lane & 15;

  const int rowBase = blockIdx.y * GBM;
  const int colBase = blockIdx.x * GBN;

  v8f acc[2][4] = {};

  auto issue_tile = [&](int k0, int buf) {
    #pragma unroll
    for (int j = 0; j < 2; j++) {            // A: 64x32 halves = 256 x 16B
      int ci  = tid + 128*j;
      int r   = ci >> 2;
      int seg = ci & 3;
      const _Float16* g = Ah + (size_t)(rowBase + r)*ldaH + k0 + seg*8;
      unsigned ldsOff = (unsigned)(uintptr_t)(&As[buf][r][seg*8]);
      asm volatile("global_load_async_to_lds_b128 %0, %1, off"
                   :: "v"(ldsOff), "v"(g) : "memory");
    }
    #pragma unroll
    for (int j = 0; j < 4; j++) {            // W: 128x32 halves = 512 x 16B
      int ci  = tid + 128*j;
      int c   = ci >> 2;
      int seg = ci & 3;
      const _Float16* g = Wt + (size_t)(colBase + c)*K + k0 + seg*8;
      unsigned ldsOff = (unsigned)(uintptr_t)(&Ws[buf][c][seg*8]);
      asm volatile("global_load_async_to_lds_b128 %0, %1, off"
                   :: "v"(ldsOff), "v"(g) : "memory");
    }
  };

  const int nk = K / GBK;
  issue_tile(0, 0);

  for (int kt = 0; kt < nk; kt++) {
    int buf = kt & 1;
    if (kt + 1 < nk) {
      issue_tile((kt+1)*GBK, buf ^ 1);
      // 6 copies just issued for tile kt+1; async loads complete in order,
      // so <=6 outstanding means tile kt is fully resident in LDS.
      asm volatile("s_wait_asynccnt 0x6" ::: "memory");
    } else {
      asm volatile("s_wait_asynccnt 0x0" ::: "memory");
    }
    __syncthreads();

    v16h af[2], bf[4];
    #pragma unroll
    for (int mi = 0; mi < 2; mi++) {      // A frag 16x32 (ISA layout)
      int tr = wm*32 + mi*16 + l16;
      #pragma unroll
      for (int i = 0; i < 16; i++) {
        int k = (i & 7) + ((i & 8) << 1) + hi*8;  // two contiguous 8-half runs
        af[mi][i] = As[buf][tr][k];
      }
    }
    #pragma unroll
    for (int ni = 0; ni < 4; ni++) {      // B frag 32x16: two contiguous 8-half runs
      int tc = wn*64 + ni*16 + l16;
      #pragma unroll
      for (int i = 0; i < 16; i++)
        bf[ni][i] = Ws[buf][tc][(i & 7) + ((i & 8) << 1) + hi*8];
    }
    #pragma unroll
    for (int mi = 0; mi < 2; mi++)
      #pragma unroll
      for (int ni = 0; ni < 4; ni++)
        acc[mi][ni] = __builtin_amdgcn_wmma_f32_16x16x32_f16(
            false, af[mi], false, bf[ni], (short)0, acc[mi][ni], false, false);
    __syncthreads();
  }

  // epilogue: lane -> N=l16, vgpr r -> M = r + hi*8; wave-uniform col guard
  #pragma unroll
  for (int mi = 0; mi < 2; mi++) {
    #pragma unroll
    for (int ni = 0; ni < 4; ni++) {
      int colSeg = colBase + wn*64 + ni*16;
      if (colSeg >= N) continue;                 // uniform (N % 16 == 0)
      int colG = colSeg + l16;
      float bv = bias ? bias[colG] : 0.0f;
      #pragma unroll
      for (int r = 0; r < 8; r++) {
        int rowG = rowBase + wm*32 + mi*16 + hi*8 + r;
        float v = acc[mi][ni][r] + bv;
        if (act == 1)      v = fmaxf(v, 0.0f);
        else if (act == 2) v = (v > 20.0f) ? v : log1pf(__expf(v));
        if (C) {
          size_t o = (size_t)rowG*ldc + colG;
          if (accum) C[o] += v; else C[o] = v;
        }
        if (Ch) Ch[(size_t)rowG*ldh + colG] = (_Float16)v;
      }
    }
  }
}

// ---------------------------------------------------------------------------
// Causal depthwise conv (width 2) + SiLU; writes f32 (scan input) + f16 (GEMM).
// ---------------------------------------------------------------------------
__global__ __launch_bounds__(256)
void conv_silu_kernel(const float* __restrict__ uPre,  // [BN,1024]
                      const float* __restrict__ cw, const float* __restrict__ cb,
                      float* __restrict__ out, _Float16* __restrict__ outh, int dir)
{
  size_t idx = (size_t)blockIdx.x*256 + threadIdx.x;
  if (idx >= (size_t)BNROWS*DINNER) return;
  int e = (int)(idx & 511);
  size_t rowIdx = idx >> 9;
  int p = (int)(rowIdx % NVARS);
  size_t bBase = rowIdx - p;
  bool hasPrev = dir ? (p < NVARS-1) : (p > 0);
  int  pPrev   = dir ? (p + 1) : (p - 1);
  float cur  = uPre[rowIdx*1024 + e];
  float prev = hasPrev ? uPre[(bBase + (size_t)pPrev)*1024 + e] : 0.0f;
  float v = siluf_(cw[e*2+0]*prev + cw[e*2+1]*cur + cb[e]);
  out [idx] = v;
  outh[idx] = (_Float16)v;
}

// ---------------------------------------------------------------------------
// Selective scan, thread owns (b,e); 32-state in VGPRs, shfl-broadcast B/C.
// ---------------------------------------------------------------------------
__global__ __launch_bounds__(256)
void scan_kernel(float* __restrict__ dy, const float* __restrict__ u,
                 const float* __restrict__ dBC, const float* __restrict__ A_log,
                 const float* __restrict__ Dp, int dir)
{
  int e    = blockIdx.x*256 + threadIdx.x;
  int b    = blockIdx.y;
  int lane = threadIdx.x & 31;

  float Arow[32];
  #pragma unroll
  for (int s = 0; s < 32; s++) Arow[s] = -__expf(A_log[(size_t)e*32 + s]);
  float Dval = Dp[e];
  float h[32];
  #pragma unroll
  for (int s = 0; s < 32; s++) h[s] = 0.0f;

  size_t bBase = (size_t)b*NVARS;
  for (int t = 0; t < NVARS; t++) {
    int p = dir ? (NVARS-1-t) : t;
    size_t row = bBase + p;
    float d    = dy[row*512 + e];
    float uu   = u [row*512 + e];
    float Breg = dBC[row*96 + 32 + lane];
    float Creg = dBC[row*96 + 64 + lane];
    if (t + 1 < NVARS) {
      int pn = dir ? (NVARS-2-t) : (t+1);
      __builtin_prefetch(&dBC[(bBase + pn)*96 + 32 + lane], 0, 3);
    }
    float du  = d*uu;
    float acc = 0.0f;
    #pragma unroll
    for (int s = 0; s < 32; s++) {
      float bs = __shfl(Breg, s, 32);
      float cs = __shfl(Creg, s, 32);
      h[s] = __expf(d*Arow[s])*h[s] + du*bs;
      acc  = fmaf(h[s], cs, acc);
    }
    dy[row*512 + e] = acc + Dval*uu;
  }
}

// ---------------------------------------------------------------------------
// g = y * silu(z) -> f16 only (consumed by out_proj GEMM)
// ---------------------------------------------------------------------------
__global__ __launch_bounds__(256)
void gate_kernel(const float* __restrict__ y, const float* __restrict__ uz,
                 _Float16* __restrict__ outh)
{
  size_t idx = (size_t)blockIdx.x*256 + threadIdx.x;
  if (idx >= (size_t)BNROWS*DINNER) return;
  int e = (int)(idx & 511);
  size_t row = idx >> 9;
  float z = uz[row*1024 + 512 + e];
  outh[idx] = (_Float16)(y[idx] * siluf_(z));
}

// ---------------------------------------------------------------------------
// LayerNorm over 512, optional residual; dual f32 + f16 outputs.
// ---------------------------------------------------------------------------
__global__ __launch_bounds__(256)
void ln_kernel(const float* __restrict__ x, const float* __restrict__ res,
               const float* __restrict__ gw, const float* __restrict__ gb,
               float* __restrict__ out, _Float16* __restrict__ outh)
{
  int row  = blockIdx.x*8 + (threadIdx.x>>5);
  int lane = threadIdx.x & 31;
  if (row >= BNROWS) return;
  const float* xr = x + (size_t)row*512;
  const float* rr = res ? res + (size_t)row*512 : nullptr;
  float v[16]; float s = 0.0f;
  #pragma unroll
  for (int j = 0; j < 16; j++) {
    float t = xr[j*32 + lane];
    if (rr) t += rr[j*32 + lane];
    v[j] = t; s += t;
  }
  for (int off=16; off; off>>=1) s += __shfl_xor(s, off, 32);
  float mean = s * (1.0f/512.0f);
  float q = 0.0f;
  #pragma unroll
  for (int j = 0; j < 16; j++) { float d = v[j]-mean; q += d*d; }
  for (int off=16; off; off>>=1) q += __shfl_xor(q, off, 32);
  float rstd = rsqrtf(q*(1.0f/512.0f) + EPSF);
  float*    orow = out  + (size_t)row*512;
  _Float16* hrow = outh + (size_t)row*512;
  #pragma unroll
  for (int j = 0; j < 16; j++) {
    int c = j*32 + lane;
    float o = (v[j]-mean)*rstd*gw[c] + gb[c];
    orow[c] = o;
    hrow[c] = (_Float16)o;
  }
}

// ---------------------------------------------------------------------------
__global__ __launch_bounds__(256)
void out_kernel(const float* __restrict__ dec,
                const float* __restrict__ means, const float* __restrict__ stdev,
                float* __restrict__ out)
{
  size_t idx = (size_t)blockIdx.x*256 + threadIdx.x;
  if (idx >= (size_t)NBATCH*96*NVARS) return;
  int n = (int)(idx % NVARS);
  size_t r = idx / NVARS;
  int l = (int)(r % 96);
  int b = (int)(r / 96);
  size_t bn = (size_t)b*NVARS + n;
  out[idx] = dec[bn*96 + l]*stdev[bn] + means[bn];
}

// ===========================================================================
extern "C" void kernel_launch(void* const* d_in, const int* in_sizes, int n_in,
                              void* d_out, int out_size, void* d_ws, size_t ws_size,
                              hipStream_t stream)
{
  (void)in_sizes; (void)n_in; (void)out_size; (void)ws_size;

  const float* hist    = (const float*)d_in[0];
  const float* emb_w   = (const float*)d_in[1];
  const float* emb_b   = (const float*)d_in[2];
  const float* m_in_w  = (const float*)d_in[3];
  const float* m_conv_w= (const float*)d_in[4];
  const float* m_conv_b= (const float*)d_in[5];
  const float* m_x_w   = (const float*)d_in[6];
  const float* m_dt_w  = (const float*)d_in[7];
  const float* m_dt_b  = (const float*)d_in[8];
  const float* m_A_log = (const float*)d_in[9];
  const float* m_D     = (const float*)d_in[10];
  const float* m_out_w = (const float*)d_in[11];
  const float* ffn_w1  = (const float*)d_in[12];
  const float* ffn_b1  = (const float*)d_in[13];
  const float* ffn_w2  = (const float*)d_in[14];
  const float* ffn_b2  = (const float*)d_in[15];
  const float* norm1_w = (const float*)d_in[16];
  const float* norm1_b = (const float*)d_in[17];
  const float* norm2_w = (const float*)d_in[18];
  const float* norm2_b = (const float*)d_in[19];
  const float* normf_w = (const float*)d_in[20];
  const float* normf_b = (const float*)d_in[21];
  const float* proj_w  = (const float*)d_in[22];
  const float* proj_b  = (const float*)d_in[23];
  float* out = (float*)d_out;

  const size_t BNr = (size_t)BNROWS;
  float* S_x   = (float*)d_ws;                 // [BN,512]  residual stream (f32)
  float* S_acc = S_x   + BNr*512;              // [BN,512]  fwd+rev accumulation
  float* S_a   = S_acc + BNr*512;              // [BN,1024] in_proj out (u|z)
  float* S_b   = S_a   + BNr*1024;             // [BN,512]  u (conv+silu) / ffn2 out
  float* S_c   = S_b   + BNr*512;              // [BN,512]  delta -> y (in place)
  float* S_d   = S_c   + BNr*512;              // [BN,96]   dBC / dec
  float* means = S_d   + BNr*96;
  float* stdev = means + BNr;
  _Float16* H_x = (_Float16*)(stdev + BNr);    // f16 of residual x
  _Float16* H_b = H_x + BNr*512;               // f16 of conv out / lnf out
  _Float16* H_c = H_b + BNr*512;               // f16 of gated y
  _Float16* H_a = H_c + BNr*512;               // f16 FFN hidden
  _Float16* H_d = H_a + BNr*512;               // f16 [BN,96]: xn / dBC
  _Float16* Wt  = H_d + BNr*96;                // f16 [<=1024,512] packed W^T

  const int EW_BLOCKS  = (int)((BNr*512 + 255)/256);
  const int RW_BLOCKS  = (BNROWS + 7)/8;
  const int OUT_BLOCKS = (int)(((size_t)NBATCH*96*NVARS + 255)/256);

  auto gemm = [&](const _Float16* Ah, int ldaH, int K, const float* W, int N,
                  const float* bias, float* C, int ldc,
                  _Float16* Ch, int act, int accum) {
    int Npad = ((N + GBN - 1)/GBN)*GBN;
    size_t nw = (size_t)Npad*(size_t)K;
    pack_w_kernel<<<dim3((unsigned)((nw+255)/256)), 256, 0, stream>>>(W, N, K, Wt, Npad);
    dim3 grid(Npad/GBN, BNROWS/GBM);
    gemm_kernel<<<grid, dim3(128), 0, stream>>>(Ah, ldaH, Wt, bias, C, ldc,
                                                Ch, N, N, K, act, accum);
  };

  // --- normalize + inverted embedding ---
  stats_kernel<<<RW_BLOCKS, 256, 0, stream>>>(hist, H_d, means, stdev);
  gemm(H_d, 96, 96, emb_w, DMODEL, emb_b, S_x, DMODEL, H_x, 0, 0);

  for (int l = 0; l < 2; l++) {
    for (int dir = 0; dir < 2; dir++) {
      size_t md = (size_t)(l*2 + dir);
      const float* in_w  = m_in_w   + md*(size_t)DMODEL*1024;
      const float* cw    = m_conv_w + md*(size_t)DINNER*2;
      const float* cb    = m_conv_b + md*(size_t)DINNER;
      const float* x_w   = m_x_w    + md*(size_t)DINNER*96;
      const float* dt_w  = m_dt_w   + md*(size_t)DTRANK*DINNER;
      const float* dt_b  = m_dt_b   + md*(size_t)DINNER;
      const float* A_log = m_A_log  + md*(size_t)DINNER*DSTATE;
      const float* Dp    = m_D      + md*(size_t)DINNER;
      const float* out_w = m_out_w  + md*(size_t)DINNER*DMODEL;

      gemm(H_x, 512, 512, in_w, 1024, nullptr, S_a, 1024, nullptr, 0, 0);
      conv_silu_kernel<<<EW_BLOCKS, 256, 0, stream>>>(S_a, cw, cb, S_b, H_b, dir);
      gemm(H_b, 512, 512, x_w, 96, nullptr, S_d, 96, H_d, 0, 0);
      gemm(H_d, 96, 32, dt_w, 512, dt_b, S_c, 512, nullptr, 2, 0);
      scan_kernel<<<dim3(2, NBATCH), 256, 0, stream>>>(S_c, S_b, S_d, A_log, Dp, dir);
      gate_kernel<<<EW_BLOCKS, 256, 0, stream>>>(S_c, S_a, H_c);
      gemm(H_c, 512, 512, out_w, 512, nullptr, S_acc, 512, nullptr, 0, dir);
    }
    ln_kernel<<<RW_BLOCKS, 256, 0, stream>>>(S_x, S_acc,
                                             norm1_w + (size_t)l*512,
                                             norm1_b + (size_t)l*512, S_x, H_x);
    gemm(H_x, 512, 512, ffn_w1 + (size_t)l*512*512, 512,
         ffn_b1 + (size_t)l*512, nullptr, 0, H_a, 1, 0);
    gemm(H_a, 512, 512, ffn_w2 + (size_t)l*512*512, 512,
         ffn_b2 + (size_t)l*512, S_b, 512, nullptr, 0, 0);
    ln_kernel<<<RW_BLOCKS, 256, 0, stream>>>(S_x, S_b,
                                             norm2_w + (size_t)l*512,
                                             norm2_b + (size_t)l*512, S_x, H_x);
  }

  ln_kernel<<<RW_BLOCKS, 256, 0, stream>>>(S_x, nullptr, normf_w, normf_b, S_b, H_b);
  gemm(H_b, 512, 512, proj_w, 96, proj_b, S_d, 96, nullptr, 0, 0);
  out_kernel<<<OUT_BLOCKS, 256, 0, stream>>>(S_d, means, stdev, out);
}
